// GraphAttentionLayer_54159537602884
// MI455X (gfx1250) — matrix operand order repaired and checked by reading, not compiled, not run on hardware
//
#include <hip/hip_runtime.h>

typedef __attribute__((ext_vector_type(16))) _Float16 v16h;
typedef __attribute__((ext_vector_type(8)))  _Float16 h8;
typedef __attribute__((ext_vector_type(8)))  float    v8f;

namespace {
constexpr int kB    = 4;
constexpr int kN    = 512;
constexpr int kInF  = 256;
constexpr int kOutF = 128;
constexpr int kHalfF = 64;
constexpr int kEdgeF = 16;
constexpr float kAlpha = 0.2f;
}

__device__ __forceinline__ v16h cat8(h8 lo, h8 hi) {
  return __builtin_shufflevector(lo, hi, 0,1,2,3,4,5,6,7,8,9,10,11,12,13,14,15);
}

__device__ __forceinline__ v8f wmma16(v16h a, v16h b, v8f c) {
  // D = A(16x32 f16) * B(32x16 f16) + C(16x16 f32)
  return __builtin_amdgcn_wmma_f32_16x16x32_f16(false, a, false, b, (short)0, c, false, false);
}

__device__ __forceinline__ v16h pack16(float4 a0, float4 a1, float4 a2, float4 a3) {
  v16h r;
  r[0]=(_Float16)a0.x;  r[1]=(_Float16)a0.y;  r[2]=(_Float16)a0.z;  r[3]=(_Float16)a0.w;
  r[4]=(_Float16)a1.x;  r[5]=(_Float16)a1.y;  r[6]=(_Float16)a1.z;  r[7]=(_Float16)a1.w;
  r[8]=(_Float16)a2.x;  r[9]=(_Float16)a2.y;  r[10]=(_Float16)a2.z; r[11]=(_Float16)a2.w;
  r[12]=(_Float16)a3.x; r[13]=(_Float16)a3.y; r[14]=(_Float16)a3.z; r[15]=(_Float16)a3.w;
  return r;
}

__device__ __forceinline__ v16h pack8z(float4 a0, float4 a1) {
  v16h r = {};
  r[0]=(_Float16)a0.x; r[1]=(_Float16)a0.y; r[2]=(_Float16)a0.z; r[3]=(_Float16)a0.w;
  r[4]=(_Float16)a1.x; r[5]=(_Float16)a1.y; r[6]=(_Float16)a1.z; r[7]=(_Float16)a1.w;
  return r;
}

// ---------------------------------------------------------------------------
// Kernel 1: Wh = h @ W  (M=2048, K=256, N=128) via WMMA; also Wh1/Wh2 = Wh @ a
// One wave per 16-row M-tile. 16 blocks x 256 threads = 128 waves.
// ---------------------------------------------------------------------------
__global__ __launch_bounds__(256)
void wh_kernel(const float* __restrict__ h, const float* __restrict__ W,
               const float* __restrict__ a,
               float* __restrict__ Wh, float* __restrict__ Wh1,
               float* __restrict__ Wh2) {
  const int lane = threadIdx.x & 31;
  const int wave = threadIdx.x >> 5;
  const int m    = lane & 15;
  const int hh   = lane >> 4;
  const int mtile = blockIdx.x * 8 + wave;   // 0..127
  const int r0 = mtile * 16;

  v8f zero = {};
  v8f acc[8];
#pragma unroll
  for (int t = 0; t < 8; ++t) acc[t] = zero;

  const float* hp = h + (size_t)(r0 + m) * kInF;

  for (int c = 0; c < 8; ++c) {
    const int kb = c * 32;
    float4 a0 = *(const float4*)(hp + kb + 8 * hh);
    float4 a1 = *(const float4*)(hp + kb + 8 * hh + 4);
    float4 a2 = *(const float4*)(hp + kb + 16 + 8 * hh);
    float4 a3 = *(const float4*)(hp + kb + 16 + 8 * hh + 4);
    v16h af = pack16(a0, a1, a2, a3);
#pragma unroll
    for (int t = 0; t < 8; ++t) {
      const int n = t * 16 + m;
      v16h bf;
#pragma unroll
      for (int k = 0; k < 16; ++k)
        bf[k] = (_Float16)W[(size_t)(kb + 8 * hh + (k & 7) + ((k >> 3) << 4)) * kOutF + n];
      acc[t] = wmma16(af, bf, acc[t]);
    }
  }

  // a-projections: per-lane partials, then cross-lane (width 16) reduction
  float p1[8], p2[8];
#pragma unroll
  for (int v = 0; v < 8; ++v) { p1[v] = 0.0f; p2[v] = 0.0f; }
#pragma unroll
  for (int t = 0; t < 8; ++t) {
    const float a1v = a[t * 16 + m];
    const float a2v = a[kOutF + t * 16 + m];
#pragma unroll
    for (int v = 0; v < 8; ++v) {
      p1[v] += acc[t][v] * a1v;
      p2[v] += acc[t][v] * a2v;
    }
  }
#pragma unroll
  for (int v = 0; v < 8; ++v)
#pragma unroll
    for (int off = 1; off < 16; off <<= 1) {
      p1[v] += __shfl_xor(p1[v], off, 16);
      p2[v] += __shfl_xor(p2[v], off, 16);
    }

#pragma unroll
  for (int t = 0; t < 8; ++t)
#pragma unroll
    for (int v = 0; v < 8; ++v)
      Wh[(size_t)(r0 + v + 8 * hh) * kOutF + t * 16 + m] = acc[t][v];

  if (m == 0) {
#pragma unroll
    for (int v = 0; v < 8; ++v) {
      Wh1[r0 + v + 8 * hh] = p1[v];
      Wh2[r0 + v + 8 * hh] = p2[v];
    }
  }
}

// ---------------------------------------------------------------------------
// Kernel 2: fused edge-FFN (WMMA) + attention logits + softmax + aggregation.
// One block per (b,i); 8 waves, each owning 4 j-tiles of 16.
// ---------------------------------------------------------------------------
__global__ __launch_bounds__(256)
void gat_kernel(const float* __restrict__ edge, const int* __restrict__ adj,
                const float* __restrict__ x,
                const float* __restrict__ w1, const float* __restrict__ b1,
                const float* __restrict__ w2, const float* __restrict__ b2,
                const float* __restrict__ Wh, const float* __restrict__ Wh1,
                const float* __restrict__ Wh2,
                float* __restrict__ out_h, float* __restrict__ out_ew) {
  __shared__ __align__(16) _Float16 w1T[128][32];       // [n][k], k>=16 zero
  __shared__ __align__(16) _Float16 w2T[64][136];       // [n][k] padded rows
  __shared__ __align__(16) _Float16 ehs[8][16][136];    // per-wave eh tile
  __shared__ float att[512];
  __shared__ float red[256];

  const int tid  = threadIdx.x;
  const int lane = tid & 31;
  const int wave = tid >> 5;
  const int m    = lane & 15;
  const int hh   = lane >> 4;
  const int bi = blockIdx.x;
  const int b  = bi >> 9;
  const int i  = bi & 511;

  // Stage weights transposed as f16 (coalesced global reads)
  for (int idx = tid; idx < 128 * 32; idx += 256) {
    const int n = idx & 127, k = idx >> 7;
    w1T[n][k] = (k < kEdgeF) ? (_Float16)w1[k * kOutF + n] : (_Float16)0.0f;
  }
  for (int idx = tid; idx < 64 * 128; idx += 256) {
    const int n = idx & 63, k = idx >> 6;
    w2T[n][k] = (_Float16)w2[k * kHalfF + n];
  }
  __syncthreads();

  float b1v[8], b2v[4];
#pragma unroll
  for (int t = 0; t < 8; ++t) b1v[t] = b1[t * 16 + m];
#pragma unroll
  for (int t = 0; t < 4; ++t) b2v[t] = b2[t * 16 + m];
  const float wh1i = Wh1[b * kN + i];
  const v8f zero = {};

  for (int jt = wave; jt < 32; jt += 8) {
    const int j0 = jt * 16;
    const float* ep =
        edge + ((size_t)(b * kN + i) * kN + (j0 + m)) * kEdgeF + 8 * hh;
    __builtin_prefetch(ep + 8 * 16 * kEdgeF, 0, 0);  // next j-tile of this wave
    float4 e0 = *(const float4*)(ep);
    float4 e1 = *(const float4*)(ep + 4);
    v16h af = pack8z(e0, e1);  // K=16 real, upper half zero-padded

    // Layer 1: [16x16] edge-tile  x  [16x128] w1  -> eh [16x128]
    v8f acc1[8];
#pragma unroll
    for (int t = 0; t < 8; ++t) acc1[t] = zero;
#pragma unroll
    for (int t = 0; t < 8; ++t) {
      const _Float16* bp = &w1T[t * 16 + m][8 * hh];
      v16h bf = cat8(*(const h8*)bp, *(const h8*)(bp + 16));
      acc1[t] = wmma16(af, bf, acc1[t]);
    }
    // ReLU(+bias) -> f16 tile in per-wave LDS
#pragma unroll
    for (int t = 0; t < 8; ++t)
#pragma unroll
      for (int v = 0; v < 8; ++v) {
        float val = acc1[t][v] + b1v[t];
        val = val > 0.0f ? val : 0.0f;
        ehs[wave][v + 8 * hh][t * 16 + m] = (_Float16)val;
      }
    __builtin_amdgcn_wave_barrier();
    asm volatile("s_wait_dscnt 0" ::: "memory");  // cross-lane LDS hazard
    __builtin_amdgcn_wave_barrier();

    // Layer 2: eh [16x128] x w2 [128x64] -> edge_w [16x64]
    v8f acc2[4];
#pragma unroll
    for (int t = 0; t < 4; ++t) acc2[t] = zero;
#pragma unroll
    for (int c = 0; c < 4; ++c) {
      const _Float16* ap = &ehs[wave][m][c * 32 + 8 * hh];
      v16h af2 = cat8(*(const h8*)ap, *(const h8*)(ap + 16));
#pragma unroll
      for (int t2 = 0; t2 < 4; ++t2) {
        const _Float16* bp = &w2T[t2 * 16 + m][c * 32 + 8 * hh];
        v16h bf = cat8(*(const h8*)bp, *(const h8*)(bp + 16));
        acc2[t2] = wmma16(af2, bf, acc2[t2]);
      }
    }

    // ReLU(+bias), stream edge_w to output, accumulate row sums
    float p[8];
#pragma unroll
    for (int v = 0; v < 8; ++v) p[v] = 0.0f;
    float* ew = out_ew + ((size_t)(b * kN + i) * kN + j0) * kHalfF;
#pragma unroll
    for (int t2 = 0; t2 < 4; ++t2)
#pragma unroll
      for (int v = 0; v < 8; ++v) {
        float val = acc2[t2][v] + b2v[t2];
        val = val > 0.0f ? val : 0.0f;
        ew[(size_t)(v + 8 * hh) * kHalfF + t2 * 16 + m] = val;
        p[v] += val;
      }
#pragma unroll
    for (int v = 0; v < 8; ++v)
#pragma unroll
      for (int off = 1; off < 16; off <<= 1)
        p[v] += __shfl_xor(p[v], off, 16);

    if (m == 0) {
#pragma unroll
      for (int v = 0; v < 8; ++v) {
        const int j = j0 + v + 8 * hh;
        float ev = wh1i + Wh2[b * kN + j];
        ev = ev > 0.0f ? ev : kAlpha * ev;            // LeakyReLU
        att[j] = ev * (64.0f + p[v]);                 // e * sum(1 + edge_w)
      }
    }
  }
  __syncthreads();

  // att / (|max| + 1e-6)
  red[tid] = fmaxf(att[tid], att[tid + 256]);
  __syncthreads();
  for (int s = 128; s > 0; s >>= 1) {
    if (tid < s) red[tid] = fmaxf(red[tid], red[tid + s]);
    __syncthreads();
  }
  const float inv = 1.0f / (fabsf(red[0]) + 1e-6f);
  __syncthreads();

  // adjacency mask, then stable softmax over j
  const int* adjr = adj + (size_t)(b * kN + i) * kN;
  float v0 = att[tid] * inv;
  float v1 = att[tid + 256] * inv;
  v0 = (adjr[tid] > 0) ? v0 : -1e6f;
  v1 = (adjr[tid + 256] > 0) ? v1 : -1e6f;

  red[tid] = fmaxf(v0, v1);
  __syncthreads();
  for (int s = 128; s > 0; s >>= 1) {
    if (tid < s) red[tid] = fmaxf(red[tid], red[tid + s]);
    __syncthreads();
  }
  const float m2 = red[0];
  __syncthreads();

  const float ex0 = __expf(v0 - m2);
  const float ex1 = __expf(v1 - m2);
  red[tid] = ex0 + ex1;
  __syncthreads();
  for (int s = 128; s > 0; s >>= 1) {
    if (tid < s) red[tid] += red[tid + s];
    __syncthreads();
  }
  const float rs = 1.0f / red[0];
  att[tid] = ex0 * rs;
  att[tid + 256] = ex1 * rs;
  __syncthreads();

  // h_prime[i,:] = attention . Wh (+x), ELU.  Wh is L2-resident (1 MB).
  const int d = tid & 127;
  const int half = tid >> 7;
  float s = 0.0f;
  const float* whb = Wh + (size_t)(b * kN) * kOutF + d;
  for (int j = half * 256; j < half * 256 + 256; ++j)
    s += att[j] * whb[(size_t)j * kOutF];
  if (half) red[d] = s;
  __syncthreads();
  if (!half) {
    float tot = s + red[d] + x[(size_t)(b * kN + i) * kOutF + d];
    tot = tot > 0.0f ? tot : (__expf(tot) - 1.0f);   // ELU
    out_h[(size_t)(b * kN + i) * kOutF + d] = tot;
  }
}

extern "C" void kernel_launch(void* const* d_in, const int* in_sizes, int n_in,
                              void* d_out, int out_size, void* d_ws,
                              size_t ws_size, hipStream_t stream) {
  const float* h    = (const float*)d_in[0];
  const int*   adj  = (const int*)d_in[1];
  const float* edge = (const float*)d_in[2];
  const float* x    = (const float*)d_in[3];
  const float* W    = (const float*)d_in[4];
  const float* a    = (const float*)d_in[5];
  const float* w1   = (const float*)d_in[6];
  const float* b1   = (const float*)d_in[7];
  const float* w2   = (const float*)d_in[8];
  const float* b2   = (const float*)d_in[9];

  float* out = (float*)d_out;
  float* Wh  = (float*)d_ws;                         // [2048,128] f32
  float* Wh1 = Wh + (size_t)kB * kN * kOutF;         // [2048]
  float* Wh2 = Wh1 + (size_t)kB * kN;                // [2048]

  wh_kernel<<<16, 256, 0, stream>>>(h, W, a, Wh, Wh1, Wh2);
  gat_kernel<<<kB * kN, 256, 0, stream>>>(
      edge, adj, x, w1, b1, w2, b2, Wh, Wh1, Wh2,
      out, out + (size_t)kB * kN * kOutF);
}
